// TempVST_9363028705447
// MI455X (gfx1250) — matrix operand order repaired and checked by reading, not compiled, not run on hardware
//
#include <hip/hip_runtime.h>
#include <math.h>

// ---- model dims ----
#define Bk   16
#define Fk   5
#define Pk   196
#define Dk   512
#define Hk   8
#define DHk  64
#define DEPTHk 12
#define PDk  384
#define NTOK (Bk * Fk * Pk)          // 15680 tokens
#define QKVW (3 * Hk * DHk)          // 1536
#define FFW  4096
#define FFH  2048
#define TCH  3920                    // ff token chunk (4 chunks)

typedef __attribute__((ext_vector_type(16))) _Float16 v16h;
typedef __attribute__((ext_vector_type(8)))  _Float16 v8h;
typedef __attribute__((ext_vector_type(2)))  _Float16 v2h;
typedef __attribute__((ext_vector_type(2)))  __fp16   v2hf;
typedef __attribute__((ext_vector_type(8)))  float    v8f;

// packed f32x2 -> f16x2 (v_cvt_pk_rtz_f16_f32), retyped to _Float16 pair
static __device__ __forceinline__ v2h pk16(float x, float y)
{
    v2hf t = __builtin_amdgcn_cvt_pkrtz(x, y);
    v2h r;
    __builtin_memcpy(&r, &t, sizeof(r));
    return r;
}

// ------------------------------------------------------------------
// Tiled WMMA GEMM:  C[M,N] = A[M,K](f32) * B[K,N](f32) (+bias) (+Cin)
// block = 256 thr (8 waves); block tile 128x128; wave tile 32x64.
// A,B staged through LDS with packed f32->f16 conversion; f32 accum.
// Requires N % 128 == 0, K % 32 == 0 (true for all calls here).
// ------------------------------------------------------------------
__global__ __launch_bounds__(256) void gemm_wmma_f16(
    const float* __restrict__ A, const float* __restrict__ B,
    const float* __restrict__ bias, const float* __restrict__ Cin,
    float* __restrict__ C, int M, int N, int K)
{
    __shared__ _Float16 sA[128][32];   // [m][k]   8 KB
    __shared__ _Float16 sB[128][32];   // [n][k]   8 KB (K-transposed)

    const int tid  = threadIdx.x;
    const int lane = tid & 31;
    const int wid  = tid >> 5;
    const int wm   = (wid & 3) << 5;          // wave m offset: 0..96
    const int wn   = (wid >> 2) << 6;         // wave n offset: 0 or 64
    const int mb   = blockIdx.x * 128;
    const int nb   = blockIdx.y * 128;
    const int lh   = lane & 15;
    const bool hiHalf = lane >= 16;

    // A staging coords: one thread owns 16 contiguous floats of one row
    const int arow = tid >> 1;
    const int acb  = (tid & 1) << 4;          // col base 0 or 16
    const int agr  = mb + arow;
    const bool aok = (agr < M);
    const float* aPtr = A + (size_t)agr * K + acb;

    v8f acc[2][4];
#pragma unroll
    for (int i = 0; i < 2; ++i)
#pragma unroll
        for (int j = 0; j < 4; ++j)
#pragma unroll
            for (int r = 0; r < 8; ++r) acc[i][j][r] = 0.0f;

    for (int k0 = 0; k0 < K; k0 += 32) {
        // ---- stage A tile: 4x global_load_b128, pack, 2x ds_store_b128
        float4 f0, f1, f2, f3;
        if (aok) {
            const float4* pA = (const float4*)(aPtr + k0);
            f0 = pA[0]; f1 = pA[1]; f2 = pA[2]; f3 = pA[3];
        } else {
            f0 = make_float4(0.f, 0.f, 0.f, 0.f);
            f1 = f0; f2 = f0; f3 = f0;
        }
        {
            v2h p0 = pk16(f0.x, f0.y);
            v2h p1 = pk16(f0.z, f0.w);
            v2h p2 = pk16(f1.x, f1.y);
            v2h p3 = pk16(f1.z, f1.w);
            v2h p4 = pk16(f2.x, f2.y);
            v2h p5 = pk16(f2.z, f2.w);
            v2h p6 = pk16(f3.x, f3.y);
            v2h p7 = pk16(f3.z, f3.w);
            v8h h0 = {p0[0], p0[1], p1[0], p1[1], p2[0], p2[1], p3[0], p3[1]};
            v8h h1 = {p4[0], p4[1], p5[0], p5[1], p6[0], p6[1], p7[0], p7[1]};
            *(v8h*)&sA[arow][acb]     = h0;
            *(v8h*)&sA[arow][acb + 8] = h1;
        }

        // ---- stage B tile transposed [n][k]: 4x float4 loads per thread
#pragma unroll
        for (int i = 0; i < 4; ++i) {
            int idx = i * 256 + tid;          // 1024 float4s = 32x128 tile
            int kk  = idx >> 5;               // 0..31
            int n4  = (idx & 31) << 2;        // 0,4,...,124
            float4 f = *(const float4*)(B + (size_t)(k0 + kk) * N + nb + n4);
            sB[n4 + 0][kk] = (_Float16)f.x;
            sB[n4 + 1][kk] = (_Float16)f.y;
            sB[n4 + 2][kk] = (_Float16)f.z;
            sB[n4 + 3][kk] = (_Float16)f.w;
        }

        // prefetch next K-chunk while this one is consumed
        if (k0 + 32 < K) {
            if (aok) __builtin_prefetch(aPtr + k0 + 32, 0, 1);
            __builtin_prefetch(B + (size_t)(k0 + 32 + (tid >> 3)) * N + nb + ((tid & 7) << 4), 0, 1);
        }
        __syncthreads();

        // A fragment per ISA 16-bit 16x32 layout:
        // lanes 0-15: halves = K0..7 | K16..23 ; lanes 16-31: K8..15 | K24..31
        v16h af[2];
#pragma unroll
        for (int mt = 0; mt < 2; ++mt) {
            int row = wm + mt * 16 + lh;
            int off = hiHalf ? 8 : 0;
            v8h lo = *(const v8h*)&sA[row][off];
            v8h hi = *(const v8h*)&sA[row][off + 16];
            af[mt] = __builtin_shufflevector(lo, hi,
                      0,1,2,3,4,5,6,7,8,9,10,11,12,13,14,15);
        }
        // B fragment: lanes 0-15 hold K0..7 | K16..23, lanes 16-31 hold K8..15 | K24..31
        v16h bf[4];
#pragma unroll
        for (int nt = 0; nt < 4; ++nt) {
            int col = wn + nt * 16 + lh;
            int off = hiHalf ? 8 : 0;
            v8h lo = *(const v8h*)&sB[col][off];
            v8h hi = *(const v8h*)&sB[col][off + 16];
            bf[nt] = __builtin_shufflevector(lo, hi,
                      0,1,2,3,4,5,6,7,8,9,10,11,12,13,14,15);
        }
#pragma unroll
        for (int mt = 0; mt < 2; ++mt)
#pragma unroll
            for (int nt = 0; nt < 4; ++nt)
                acc[mt][nt] = __builtin_amdgcn_wmma_f32_16x16x32_f16(
                    false, af[mt], false, bf[nt],
                    (short)0, acc[mt][nt], false, false);
        __syncthreads();
    }

    // epilogue: C layout VGPR r -> M = r (lanes 0-15) / 8+r (lanes 16-31)
    if (mb + 128 <= M) {
        // fast path: no row guards
#pragma unroll
        for (int mt = 0; mt < 2; ++mt) {
            int rbase = mb + wm + mt * 16 + (hiHalf ? 8 : 0);
#pragma unroll
            for (int nt = 0; nt < 4; ++nt) {
                int col = nb + wn + nt * 16 + lh;
                float bv = bias ? bias[col] : 0.0f;
#pragma unroll
                for (int r = 0; r < 8; ++r) {
                    size_t o = (size_t)(rbase + r) * N + col;
                    float v = acc[mt][nt][r] + bv;
                    if (Cin) v += Cin[o];
                    C[o] = v;
                }
            }
        }
    } else {
#pragma unroll
        for (int mt = 0; mt < 2; ++mt) {
            int rbase = mb + wm + mt * 16 + (hiHalf ? 8 : 0);
#pragma unroll
            for (int nt = 0; nt < 4; ++nt) {
                int col = nb + wn + nt * 16 + lh;
                float bv = bias ? bias[col] : 0.0f;
#pragma unroll
                for (int r = 0; r < 8; ++r) {
                    int row = rbase + r;
                    if (row < M) {
                        size_t o = (size_t)row * N + col;
                        float v = acc[mt][nt][r] + bv;
                        if (Cin) v += Cin[o];
                        C[o] = v;
                    }
                }
            }
        }
    }
}

// ------------------------------------------------------------------
// LayerNorm over D=512, one block per token row
// ------------------------------------------------------------------
__global__ __launch_bounds__(128) void ln_kernel(
    const float* __restrict__ x, float* __restrict__ y,
    const float* __restrict__ g, const float* __restrict__ b)
{
    __shared__ float s1[128], s2[128];
    const int row = blockIdx.x, tid = threadIdx.x;
    const float* xr = x + (size_t)row * Dk;
    float v[4], s = 0.0f, ss = 0.0f;
#pragma unroll
    for (int i = 0; i < 4; ++i) {
        v[i] = xr[tid + i * 128];
        s += v[i]; ss += v[i] * v[i];
    }
    s1[tid] = s; s2[tid] = ss;
    __syncthreads();
    for (int o = 64; o > 0; o >>= 1) {
        if (tid < o) { s1[tid] += s1[tid + o]; s2[tid] += s2[tid + o]; }
        __syncthreads();
    }
    float mu   = s1[0] * (1.0f / Dk);
    float var  = s2[0] * (1.0f / Dk) - mu * mu;
    float rstd = rsqrtf(var + 1e-5f);
    float* yr = y + (size_t)row * Dk;
#pragma unroll
    for (int i = 0; i < 4; ++i) {
        int c = tid + i * 128;
        yr[c] = (v[i] - mu) * rstd * g[c] + b[c];
    }
}

// x[b,n,:] += pos_emb[n,:]
__global__ void addpos_kernel(float* __restrict__ x, const float* __restrict__ pe)
{
    int idx = blockIdx.x * 256 + threadIdx.x;
    if (idx >= NTOK * Dk) return;
    x[idx] += pe[idx % (Fk * Pk * Dk)];
}

// GEGLU: gf = u * gelu_exact(g), ff is [rows,4096] -> gf [rows,2048]
__global__ void geglu_kernel(const float* __restrict__ ff,
                             float* __restrict__ gf, int rows)
{
    int idx = blockIdx.x * 256 + threadIdx.x;
    if (idx >= rows * FFH) return;
    int r = idx >> 11, c = idx & (FFH - 1);
    float u = ff[(size_t)r * FFW + c];
    float g = ff[(size_t)r * FFW + FFH + c];
    float gel = 0.5f * g * (1.0f + erff(g * 0.70710678118654752f));
    gf[idx] = u * gel;
}

// ------------------------------------------------------------------
// Axial attention, online softmax. One thread per (b, h, token).
// time axis: attend over F=5 frames (same patch); space: over P=196.
// qkv: [B, 980, 1536] = q|k|v each [H,DH]. out: [B, 980, 512]
// ------------------------------------------------------------------
__global__ __launch_bounds__(256) void axial_attn_kernel(
    const float* __restrict__ qkv, float* __restrict__ out, int time_axis)
{
    int idx = blockIdx.x * 256 + threadIdx.x;
    if (idx >= Bk * Hk * Fk * Pk) return;
    int n = idx % (Fk * Pk);
    int t = idx / (Fk * Pk);
    int h = t % Hk, b = t / Hk;
    int f = n / Pk, p = n % Pk;

    int L, base_n, stride;
    if (time_axis) { L = Fk;  base_n = p;       stride = Pk; }
    else           { L = Pk;  base_n = f * Pk;  stride = 1;  }

    const float* q = qkv + ((size_t)(b * (Fk * Pk) + n) * QKVW) + h * DHk;
    float qr[DHk];
#pragma unroll
    for (int d = 0; d < DHk; ++d) qr[d] = q[d] * 0.125f;   // DH^-0.5

    float m = -INFINITY, l = 0.0f, acc[DHk];
#pragma unroll
    for (int d = 0; d < DHk; ++d) acc[d] = 0.0f;

    for (int j = 0; j < L; ++j) {
        const float* kr = qkv + ((size_t)(b * (Fk * Pk) + base_n + j * stride) * QKVW)
                          + (Hk * DHk) + h * DHk;
        float s = 0.0f;
        for (int d = 0; d < DHk; ++d) s += qr[d] * kr[d];
        float mn = fmaxf(m, s);
        float corr = __expf(m - mn);
        float w    = __expf(s - mn);
        l = l * corr + w;
        const float* vr = kr + (Hk * DHk);
        for (int d = 0; d < DHk; ++d) acc[d] = acc[d] * corr + w * vr[d];
        m = mn;
    }
    float inv = 1.0f / l;
    float* o = out + ((size_t)(b * (Fk * Pk) + n) * Dk) + h * DHk;
#pragma unroll
    for (int d = 0; d < DHk; ++d) o[d] = acc[d] * inv;
}

// ------------------------------------------------------------------
static inline void launch_gemm(const float* A, const float* B, const float* bias,
                               const float* Cin, float* C, int M, int N, int K,
                               hipStream_t s)
{
    dim3 grid((M + 127) / 128, N / 128);
    gemm_wmma_f16<<<grid, 256, 0, s>>>(A, B, bias, Cin, C, M, N, K);
}

extern "C" void kernel_launch(void* const* d_in, const int* in_sizes, int n_in,
                              void* d_out, int out_size, void* d_ws, size_t ws_size,
                              hipStream_t stream)
{
    const float* video  = (const float*)d_in[0];
    const float* Wpe    = (const float*)d_in[1];
    const float* bpe    = (const float*)d_in[2];
    const float* pos    = (const float*)d_in[3];
    const float* ln1g   = (const float*)d_in[4];
    const float* ln1b   = (const float*)d_in[5];
    const float* Wqkv_t = (const float*)d_in[6];
    const float* Wo_t   = (const float*)d_in[7];
    const float* bo_t   = (const float*)d_in[8];
    const float* ln2g   = (const float*)d_in[9];
    const float* ln2b   = (const float*)d_in[10];
    const float* Wqkv_s = (const float*)d_in[11];
    const float* Wo_s   = (const float*)d_in[12];
    const float* bo_s   = (const float*)d_in[13];
    const float* ln3g   = (const float*)d_in[14];
    const float* ln3b   = (const float*)d_in[15];
    const float* Wff1   = (const float*)d_in[16];
    const float* bff1   = (const float*)d_in[17];
    const float* Wff2   = (const float*)d_in[18];
    const float* bff2   = (const float*)d_in[19];
    const float* Wout   = (const float*)d_in[20];
    const float* bout   = (const float*)d_in[21];

    // workspace carve-up (~257 MB)
    float* xf   = (float*)d_ws;                      // residual [15680,512]
    float* hf   = xf   + (size_t)NTOK * Dk;          // ln/attn temp [15680,512]
    float* qkvf = hf   + (size_t)NTOK * Dk;          // [15680,1536]
    float* fff  = qkvf + (size_t)NTOK * QKVW;        // ff chunk [3920,4096]
    float* gf   = fff  + (size_t)TCH * FFW;          // geglu chunk [3920,2048]

    const int elems = NTOK * Dk;
    const int attnThreads = Bk * Hk * Fk * Pk;

    // patch embedding + positional embedding
    launch_gemm(video, Wpe, bpe, nullptr, xf, NTOK, Dk, PDk, stream);
    addpos_kernel<<<(elems + 255) / 256, 256, 0, stream>>>(xf, pos);

    for (int l = 0; l < DEPTHk; ++l) {
        // ---- time attention block ----
        ln_kernel<<<NTOK, 128, 0, stream>>>(xf, hf, ln1g + l * Dk, ln1b + l * Dk);
        launch_gemm(hf, Wqkv_t + (size_t)l * Dk * QKVW, nullptr, nullptr,
                    qkvf, NTOK, QKVW, Dk, stream);
        axial_attn_kernel<<<(attnThreads + 255) / 256, 256, 0, stream>>>(qkvf, hf, 1);
        launch_gemm(hf, Wo_t + (size_t)l * Dk * Dk, bo_t + l * Dk, xf,
                    xf, NTOK, Dk, Dk, stream);

        // ---- space attention block ----
        ln_kernel<<<NTOK, 128, 0, stream>>>(xf, hf, ln2g + l * Dk, ln2b + l * Dk);
        launch_gemm(hf, Wqkv_s + (size_t)l * Dk * QKVW, nullptr, nullptr,
                    qkvf, NTOK, QKVW, Dk, stream);
        axial_attn_kernel<<<(attnThreads + 255) / 256, 256, 0, stream>>>(qkvf, hf, 0);
        launch_gemm(hf, Wo_s + (size_t)l * Dk * Dk, bo_s + l * Dk, xf,
                    xf, NTOK, Dk, Dk, stream);

        // ---- GEGLU feed-forward, token-chunked ----
        ln_kernel<<<NTOK, 128, 0, stream>>>(xf, hf, ln3g + l * Dk, ln3b + l * Dk);
        for (int c = 0; c < NTOK / TCH; ++c) {
            size_t off = (size_t)c * TCH;
            launch_gemm(hf + off * Dk, Wff1 + (size_t)l * Dk * FFW,
                        bff1 + (size_t)l * FFW, nullptr,
                        fff, TCH, FFW, Dk, stream);
            geglu_kernel<<<(TCH * FFH + 255) / 256, 256, 0, stream>>>(fff, gf, TCH);
            launch_gemm(gf, Wff2 + (size_t)l * FFH * Dk, bff2 + l * Dk,
                        xf + off * Dk, xf + off * Dk, TCH, Dk, FFH, stream);
        }
    }

    // output projection -> d_out (f32)
    launch_gemm(xf, Wout, bout, nullptr, (float*)d_out, NTOK, PDk, Dk, stream);
}
